// CausalSelfAttention_34119220199727
// MI455X (gfx1250) — matrix operand order repaired and checked by reading, not compiled, not run on hardware
//
#include <hip/hip_runtime.h>
#include <hip/hip_bf16.h>

typedef __attribute__((ext_vector_type(16))) __bf16 v16bf;
typedef __attribute__((ext_vector_type(8)))  float  v8f;
typedef __attribute__((ext_vector_type(2)))  float  vf2;
typedef __attribute__((ext_vector_type(2)))  __bf16 vbf2;

union Frag {
  v16bf v;
  uint4 q[2];
};

#define DM    1024   // d_model == d_k
#define MROWS 8192   // B*T
#define T_SEQ 2048

// Native bf16 convert (RNE in hardware).
__device__ __forceinline__ unsigned short f2bf(float f) {
  __bf16 h = (__bf16)f;
  unsigned short u;
  __builtin_memcpy(&u, &h, 2);
  return u;
}
// Packed pair convert: lowers to a single v_cvt_pk_bf16_f32.
__device__ __forceinline__ unsigned int f2bf2(float lo, float hi) {
  vf2 f = {lo, hi};
  vbf2 h = __builtin_convertvector(f, vbf2);
  unsigned int u;
  __builtin_memcpy(&u, &h, 4);
  return u;
}
__device__ __forceinline__ v8f wmma_bf16(v16bf a, v16bf b, v8f c) {
  // D = A(16x32 bf16) * B(32x16 bf16) + C(16x16 f32)
  return __builtin_amdgcn_wmma_f32_16x16x32_bf16(false, a, false, b, (short)0, c,
                                                 false, false);
}

// ---------------------------------------------------------------------------
// Kernel 1: Y = X * W for W in {Wq, Wk, Wv}. 128x128 tile / block, 8 waves.
// Software-pipelined: tile t+1 is staged global->registers while tile t's
// WMMAs run on LDS. Q pre-scaled by 1/32; V stored transposed Vt[n][m].
// ---------------------------------------------------------------------------
__global__ __launch_bounds__(256) void qkv_proj_kernel(
    const float* __restrict__ X,  const float* __restrict__ Wq,
    const float* __restrict__ Wk, const float* __restrict__ Wv,
    unsigned short* __restrict__ Qbf, unsigned short* __restrict__ Kbf,
    unsigned short* __restrict__ Vt)
{
  __shared__ unsigned short Xs[128][48];   // [m][k] row-major, pad to 48 halves
  __shared__ unsigned short Ws[128][48];   // [n][k] (W transposed), pad to 48

  const int z = blockIdx.z;
  const float* W = (z == 0) ? Wq : (z == 1) ? Wk : Wv;
  const int mBase = blockIdx.x * 128;
  const int nBase = blockIdx.y * 128;

  const int tid = threadIdx.x;
  const int wave = tid >> 5, lane = tid & 31;
  const int lane16 = lane & 15, hi = lane >> 4;
  const int mOff = (wave >> 2) * 64;   // wave row: 0 / 64
  const int nOff = (wave & 3) * 32;    // wave col: 0,32,64,96
  const int kb0 = hi ? 16 : 0;         // A-frag K base
  const int kbB = hi ? 16 : 0;         // B-frag K base

  const v8f vzero = {0.f, 0.f, 0.f, 0.f, 0.f, 0.f, 0.f, 0.f};
  v8f acc[4][2];
  for (int i = 0; i < 4; ++i)
    for (int j = 0; j < 2; ++j) acc[i][j] = vzero;

  const int xr = tid >> 1, xh = tid & 1;   // X loader: 128 rows x 2 half-rows
  const int wk = tid & 31, wn = tid >> 5;  // W loader: 32 k-rows x 8 n-chunks

  float4 xreg[4], wreg[4];                 // staging registers for next tile

  auto loadRegs = [&](int kk) {
    const float4* xs = (const float4*)(X + (size_t)(mBase + xr) * DM + kk + xh * 16);
    const float4* ws = (const float4*)(W + (size_t)(kk + wk) * DM + nBase + wn * 16);
    #pragma unroll
    for (int j = 0; j < 4; ++j) xreg[j] = xs[j];
    #pragma unroll
    for (int j = 0; j < 4; ++j) wreg[j] = ws[j];
  };
  auto storeLDS = [&]() {
    unsigned int* dst = (unsigned int*)&Xs[xr][xh * 16];
    #pragma unroll
    for (int j = 0; j < 4; ++j) {
      dst[2 * j]     = f2bf2(xreg[j].x, xreg[j].y);
      dst[2 * j + 1] = f2bf2(xreg[j].z, xreg[j].w);
    }
    #pragma unroll
    for (int j = 0; j < 4; ++j) {
      Ws[wn * 16 + 4 * j + 0][wk] = f2bf(wreg[j].x);
      Ws[wn * 16 + 4 * j + 1][wk] = f2bf(wreg[j].y);
      Ws[wn * 16 + 4 * j + 2][wk] = f2bf(wreg[j].z);
      Ws[wn * 16 + 4 * j + 3][wk] = f2bf(wreg[j].w);
    }
  };

  loadRegs(0);
  storeLDS();
  __syncthreads();

  for (int kk = 0; kk < DM; kk += 32) {
    const bool more = (kk + 32) < DM;
    if (more) loadRegs(kk + 32);   // overlap global loads with WMMA below

    Frag a[4], bfr[2];
    #pragma unroll
    for (int i = 0; i < 4; ++i) {
      const unsigned short* p = &Xs[mOff + 16 * i + lane16][kb0];
      a[i].q[0] = *(const uint4*)p;          // K kb0..kb0+7
      a[i].q[1] = *(const uint4*)(p + 8);    // K kb0+8..kb0+15
    }
    #pragma unroll
    for (int j = 0; j < 2; ++j) {
      const unsigned short* p = &Ws[nOff + 16 * j + lane16][kbB];
      bfr[j].q[0] = *(const uint4*)p;        // K kbB..kbB+7
      bfr[j].q[1] = *(const uint4*)(p + 8);  // K kbB+8..kbB+15
    }
    #pragma unroll
    for (int i = 0; i < 4; ++i)
      #pragma unroll
      for (int j = 0; j < 2; ++j)
        acc[i][j] = wmma_bf16(a[i].v, bfr[j].v, acc[i][j]);

    __syncthreads();               // all waves done reading this tile
    if (more) {
      storeLDS();                  // commit staged tile
      __syncthreads();             // tile visible before next compute
    }
  }

  const float scale = (z == 0) ? 0.03125f : 1.0f;   // fold 1/sqrt(1024) into Q
  if (z < 2) {
    unsigned short* out = (z == 0) ? Qbf : Kbf;
    for (int i = 0; i < 4; ++i) {
      const int mB = mBase + mOff + 16 * i + hi * 8;  // D layout: reg r -> row mB+r
      for (int j = 0; j < 2; ++j) {
        const int col = nBase + nOff + 16 * j + lane16;
        #pragma unroll
        for (int r = 0; r < 8; ++r)
          out[(size_t)(mB + r) * DM + col] = f2bf(acc[i][j][r] * scale);
      }
    }
  } else {
    // V transposed store: lane's 8 D-rows are contiguous along m -> one b128 store
    for (int i = 0; i < 4; ++i) {
      const int mB = mBase + mOff + 16 * i + hi * 8;
      for (int j = 0; j < 2; ++j) {
        const int col = nBase + nOff + 16 * j + lane16;
        uint4 pkd;
        pkd.x = f2bf2(acc[i][j][0], acc[i][j][1]);
        pkd.y = f2bf2(acc[i][j][2], acc[i][j][3]);
        pkd.z = f2bf2(acc[i][j][4], acc[i][j][5]);
        pkd.w = f2bf2(acc[i][j][6], acc[i][j][7]);
        *(uint4*)(Vt + (size_t)col * MROWS + mB) = pkd;
      }
    }
  }
}

// ---------------------------------------------------------------------------
// Kernel 2: causal flash attention. Block = 64 query rows, one batch, one
// 512-wide output-column half (blockIdx.z). Loops over 64-key chunks with
// online softmax. Q/K/V fragments built straight from L2-resident bf16
// buffers; only S (f32) and P (bf16) round-trip through LDS.
// ---------------------------------------------------------------------------
__global__ __launch_bounds__(256) void attn_kernel(
    const unsigned short* __restrict__ Qbf, const unsigned short* __restrict__ Kbf,
    const unsigned short* __restrict__ Vt, float* __restrict__ out)
{
  __shared__ float Slds[64][68];           // scores tile, f32
  __shared__ unsigned short Plds[64][72];  // probabilities tile, bf16
  __shared__ float red[64][4];
  __shared__ float mrow[64], lrow[64], arow[64];

  const int qt = blockIdx.x;
  const int bb = blockIdx.y;
  const int zn = blockIdx.z;               // output-column half: 0 / 1
  const int qBase = qt * 64;
  const size_t rowBase = (size_t)bb * T_SEQ;

  const int tid = threadIdx.x;
  const int wave = tid >> 5, lane = tid & 31;
  const int lane16 = lane & 15, hi = lane >> 4;
  const int kb0 = hi ? 16 : 0, kbB = hi ? 16 : 0;

  // S-phase partition: 4 waves over M (16 rows each) x 2 waves over N (32 each)
  const int smOff = (wave >> 1) * 16;
  const int snOff = (wave & 1) * 32;
  // PV-phase partition: 4 waves over M x 2 waves over N (256 cols each)
  const int pmOff = (wave & 3) * 16;
  const int pnOff = zn * 512 + (wave >> 2) * 256;

  const v8f vzero = {0.f, 0.f, 0.f, 0.f, 0.f, 0.f, 0.f, 0.f};
  v8f oacc[16];
  #pragma unroll
  for (int f = 0; f < 16; ++f) oacc[f] = vzero;

  if (tid < 64) { mrow[tid] = -__builtin_inff(); lrow[tid] = 0.f; }
  __syncthreads();

  const int row = tid >> 2, part = tid & 3;   // softmax: 4 threads per q-row

  for (int c = 0; c <= qt; ++c) {
    const int sBase = c * 64;

    // Prefetch this chunk's Vt column lines (one 128B line per column; lanes l and l+16 share it)
    // so they are resident by the time phase 4 builds B-fragments from them.
    #pragma unroll 4
    for (int nt = 0; nt < 16; ++nt) {
      const unsigned short* vp =
          Vt + (size_t)(pnOff + nt * 16 + lane16) * MROWS + rowBase + sBase;
      __builtin_prefetch(vp, 0, 0);
    }

    // ---- Phase 1: S(64x64) = Q(64x1024) * K^T, direct-from-global frags ----
    v8f sacc0 = vzero, sacc1 = vzero;
    const unsigned short* qp  = Qbf + (rowBase + qBase + smOff + lane16) * DM;
    const unsigned short* kp0 = Kbf + (rowBase + sBase + snOff + lane16) * DM;
    const unsigned short* kp1 = kp0 + 16 * DM;
    for (int kk = 0; kk < DM; kk += 32) {
      if (kk + 64 < DM) {        // pull K/Q lines two k-steps ahead toward WGP
        __builtin_prefetch(qp + kk + 64, 0, 0);
        __builtin_prefetch(kp0 + kk + 64, 0, 0);
        __builtin_prefetch(kp1 + kk + 64, 0, 0);
      }
      Frag a, b0, b1;
      a.q[0]  = *(const uint4*)(qp + kk + kb0);
      a.q[1]  = *(const uint4*)(qp + kk + kb0 + 8);
      b0.q[0] = *(const uint4*)(kp0 + kk + kbB);
      b0.q[1] = *(const uint4*)(kp0 + kk + kbB + 8);
      b1.q[0] = *(const uint4*)(kp1 + kk + kbB);
      b1.q[1] = *(const uint4*)(kp1 + kk + kbB + 8);
      sacc0 = wmma_bf16(a.v, b0.v, sacc0);
      sacc1 = wmma_bf16(a.v, b1.v, sacc1);
    }

    // ---- Phase 2: D-layout -> LDS ----
    #pragma unroll
    for (int r = 0; r < 8; ++r) {
      Slds[smOff + r + hi * 8][snOff + lane16]      = sacc0[r];
      Slds[smOff + r + hi * 8][snOff + 16 + lane16] = sacc1[r];
    }
    __syncthreads();

    // ---- Phase 3: online softmax (causal mask applied here) ----
    const int qIdx = qBase + row;
    float sv[16];
    float lmax = -__builtin_inff();
    #pragma unroll
    for (int e = 0; e < 16; ++e) {
      const int sIdx = sBase + part * 16 + e;
      float v = Slds[row][part * 16 + e];
      v = (sIdx <= qIdx) ? v : -__builtin_inff();
      sv[e] = v;
      lmax = fmaxf(lmax, v);
    }
    red[row][part] = lmax;
    __syncthreads();
    if (part == 0) {
      const float cm = fmaxf(fmaxf(red[row][0], red[row][1]),
                             fmaxf(red[row][2], red[row][3]));
      const float mo = mrow[row];
      const float mn = fmaxf(mo, cm);
      arow[row] = __expf(mo - mn);   // -inf -> 0 on first chunk
      mrow[row] = mn;
    }
    __syncthreads();
    const float mn = mrow[row];
    float ls = 0.f;
    float pv[16];
    #pragma unroll
    for (int e = 0; e < 16; ++e) {
      const float p = (sv[e] == -__builtin_inff()) ? 0.f : __expf(sv[e] - mn);
      ls += p;
      pv[e] = p;
    }
    {
      // packed bf16 pair stores: 8 dword ds_stores instead of 16 b16 stores
      unsigned int* pdst = (unsigned int*)&Plds[row][part * 16];
      #pragma unroll
      for (int e = 0; e < 8; ++e) pdst[e] = f2bf2(pv[2 * e], pv[2 * e + 1]);
    }
    red[row][part] = ls;
    __syncthreads();
    if (part == 0)
      lrow[row] = lrow[row] * arow[row] +
                  red[row][0] + red[row][1] + red[row][2] + red[row][3];
    __syncthreads();

    // ---- Phase 4: rescale O, then O += P(64x64) * V(64x512-half) ----
    float al[8];
    #pragma unroll
    for (int r = 0; r < 8; ++r) al[r] = arow[pmOff + r + hi * 8];
    #pragma unroll
    for (int f = 0; f < 16; ++f)
      #pragma unroll
      for (int r = 0; r < 8; ++r) oacc[f][r] *= al[r];

    #pragma unroll
    for (int ss = 0; ss < 64; ss += 32) {
      Frag a;
      const unsigned short* pp = &Plds[pmOff + lane16][ss + kb0];
      a.q[0] = *(const uint4*)pp;
      a.q[1] = *(const uint4*)(pp + 8);
      for (int nt = 0; nt < 16; ++nt) {
        const int col = pnOff + nt * 16 + lane16;
        const unsigned short* vp =
            Vt + (size_t)col * MROWS + rowBase + sBase + ss + kbB;
        Frag bfr;
        bfr.q[0] = *(const uint4*)vp;        // Vt[n][s] contiguous along s
        bfr.q[1] = *(const uint4*)(vp + 8);
        oacc[nt] = wmma_bf16(a.v, bfr.v, oacc[nt]);
      }
    }
  }

  // ---- Epilogue: O /= l, store f32 ----
  float linv[8];
  #pragma unroll
  for (int r = 0; r < 8; ++r) linv[r] = 1.0f / lrow[pmOff + r + hi * 8];
  for (int nt = 0; nt < 16; ++nt) {
    const int col = pnOff + nt * 16 + lane16;
    #pragma unroll
    for (int r = 0; r < 8; ++r)
      out[(rowBase + qBase + pmOff + r + hi * 8) * DM + col] =
          oacc[nt][r] * linv[r];
  }
}

// ---------------------------------------------------------------------------
extern "C" void kernel_launch(void* const* d_in, const int* in_sizes, int n_in,
                              void* d_out, int out_size, void* d_ws, size_t ws_size,
                              hipStream_t stream) {
  (void)in_sizes; (void)n_in; (void)out_size; (void)ws_size;
  const float* x  = (const float*)d_in[0];
  const float* Wq = (const float*)d_in[1];
  const float* Wk = (const float*)d_in[2];
  const float* Wv = (const float*)d_in[3];

  // Workspace layout (48 MB total): Qbf | Kbf | Vt  (each 8192*1024 bf16)
  unsigned short* Qbf = (unsigned short*)d_ws;
  unsigned short* Kbf = Qbf + (size_t)MROWS * DM;
  unsigned short* Vt  = Kbf + (size_t)MROWS * DM;
  float* out = (float*)d_out;

  qkv_proj_kernel<<<dim3(64, 8, 3), 256, 0, stream>>>(x, Wq, Wk, Wv, Qbf, Kbf, Vt);
  attn_kernel<<<dim3(32, 4, 2), 256, 0, stream>>>(Qbf, Kbf, Vt, out);
}
